// MultiHeadTimeAttention_15195594293712
// MI455X (gfx1250) — matrix operand order repaired and checked by reading, not compiled, not run on hardware
//
#include <hip/hip_runtime.h>

// ---------------------------------------------------------------------------
// Problem constants
// ---------------------------------------------------------------------------
#define B_   16
#define T_   512
#define D_   1024
#define H_   16
#define HD_  64
#define N3D_ 3072          // 3*D
#define RELW 1023          // 2*MAX_SEQ-1
#define OUT_ELEMS (B_ * T_ * D_)   // 8388608

typedef __attribute__((ext_vector_type(16))) _Float16 v16h;
typedef __attribute__((ext_vector_type(8)))  float    v8f;

// ---------------------------------------------------------------------------
// WMMA helpers (gfx1250: v_wmma_f32_16x16x32_f16, D = A(16x32) * B(32x16) + C)
// ---------------------------------------------------------------------------
__device__ inline v8f wmma_f16(v16h a, v16h b, v8f c) {
    return __builtin_amdgcn_wmma_f32_16x16x32_f16(
        /*neg_a=*/false, a, /*neg_b=*/false, b,
        /*c_mod=*/(short)0, c, /*reuse_a=*/false, /*reuse_b=*/false);
}

// A fragment: 16x32 f16, row-major source with row stride `stride` (halves).
// ISA layout: lane m=lane&15 holds row m.  half pair j: K = 2j (j<4) or 2j+8
// (j>=4), plus +8 for lanes>=16.  Pairs are K-contiguous -> b32 pair loads.
__device__ inline v16h load_frag_a16(const _Float16* __restrict__ base,
                                     int stride, int lane) {
    const _Float16* row = base + (lane & 15) * stride;
    const int kb = (lane & 16) ? 8 : 0;
    v16h a;
#pragma unroll
    for (int j = 0; j < 8; ++j) {
        const int k = ((j < 4) ? (2 * j) : (2 * j + 8)) + kb;
        a[2 * j]     = row[k];
        a[2 * j + 1] = row[k + 1];
    }
    return a;
}

// Same A fragment but from an f32 source (softmax probabilities in LDS).
__device__ inline v16h load_frag_a_f32(const float* __restrict__ base,
                                       int stride, int lane) {
    const float* row = base + (lane & 15) * stride;
    const int kb = (lane & 16) ? 8 : 0;
    v16h a;
#pragma unroll
    for (int j = 0; j < 8; ++j) {
        const int k = ((j < 4) ? (2 * j) : (2 * j + 8)) + kb;
        a[2 * j]     = (_Float16)row[k];
        a[2 * j + 1] = (_Float16)row[k + 1];
    }
    return a;
}

// B fragment: 32x16 f16.  Element (kk, n) lives at base + n*nstride + kk*kstride.
// ISA layout: lane holds column n=lane&15; half i holds K = i (i<8) or i+8 (i>=8), +8 for lanes>=16.
__device__ inline v16h load_frag_b16(const _Float16* __restrict__ base,
                                     int nstride, int kstride, int lane) {
    const _Float16* p = base + (lane & 15) * nstride +
                        ((lane & 16) ? 8 : 0) * kstride;
    v16h b;
#pragma unroll
    for (int i = 0; i < 16; ++i) b[i] = p[((i < 8) ? i : (i + 8)) * kstride];
    return b;
}

// ---------------------------------------------------------------------------
// Wave32 reductions
// ---------------------------------------------------------------------------
__device__ inline float wave_sum(float v) {
#pragma unroll
    for (int o = 16; o > 0; o >>= 1) v += __shfl_xor(v, o, 32);
    return v;
}
__device__ inline float wave_max(float v) {
#pragma unroll
    for (int o = 16; o > 0; o >>= 1) v = fmaxf(v, __shfl_xor(v, o, 32));
    return v;
}

// ---------------------------------------------------------------------------
// Kernel 1: weight conversion fp32 -> f16 (float4 vectorized)
// ---------------------------------------------------------------------------
__global__ __launch_bounds__(256) void prep_kernel(
    const float4* __restrict__ qkv_w4, const float4* __restrict__ out_w4,
    _Float16* __restrict__ qkvw_h, _Float16* __restrict__ outw_h) {
    const int i = blockIdx.x * 256 + threadIdx.x;  // float4 index
    if (i < (D_ * N3D_) / 4) {
        const float4 v = qkv_w4[i];
        _Float16* p = qkvw_h + (size_t)i * 4;
        p[0] = (_Float16)v.x; p[1] = (_Float16)v.y;
        p[2] = (_Float16)v.z; p[3] = (_Float16)v.w;
    }
    if (i < (D_ * D_) / 4) {
        const float4 v = out_w4[i];
        _Float16* p = outw_h + (size_t)i * 4;
        p[0] = (_Float16)v.x; p[1] = (_Float16)v.y;
        p[2] = (_Float16)v.z; p[3] = (_Float16)v.w;
    }
}

// ---------------------------------------------------------------------------
// Kernel 2: LayerNorm, emits f16 activations.  One block per (b,t) row.
//           float4 loads: 1 x b128 per thread.
// ---------------------------------------------------------------------------
__global__ __launch_bounds__(256) void ln_kernel(
    const float* __restrict__ x, const float* __restrict__ g,
    const float* __restrict__ bta, _Float16* __restrict__ xn) {
    const int row  = blockIdx.x;
    const int tid  = threadIdx.x;
    const int wid  = tid >> 5, lane = tid & 31;

    const float4 v = ((const float4*)(x + (size_t)row * D_))[tid];
    float s  = v.x + v.y + v.z + v.w;
    float s2 = v.x * v.x + v.y * v.y + v.z * v.z + v.w * v.w;
    s = wave_sum(s); s2 = wave_sum(s2);

    __shared__ float rs[8], rs2[8];
    if (lane == 0) { rs[wid] = s; rs2[wid] = s2; }
    __syncthreads();
    if (wid == 0) {
        float a = (lane < 8) ? rs[lane] : 0.f;
        float b = (lane < 8) ? rs2[lane] : 0.f;
        a = wave_sum(a); b = wave_sum(b);
        if (lane == 0) { rs[0] = a; rs2[0] = b; }
    }
    __syncthreads();
    const float mu  = rs[0] * (1.f / D_);
    const float var = rs2[0] * (1.f / D_) - mu * mu;
    const float rsd = rsqrtf(var + 1e-5f);

    const float4 g4 = ((const float4*)g)[tid];
    const float4 b4 = ((const float4*)bta)[tid];
    _Float16* xo = xn + (size_t)row * D_ + tid * 4;
    xo[0] = (_Float16)((v.x - mu) * rsd * g4.x + b4.x);
    xo[1] = (_Float16)((v.y - mu) * rsd * g4.y + b4.y);
    xo[2] = (_Float16)((v.z - mu) * rsd * g4.z + b4.z);
    xo[3] = (_Float16)((v.w - mu) * rsd * g4.w + b4.w);
}

// ---------------------------------------------------------------------------
// GEMM tile staging helpers (b128 global loads)
//   sA: [64][40]  (stride 40 halves = 80B, multiple of 16B -> b128 stores)
//   sB: [256][34] (transposed; 16-bit scatter stores, dword fragment reads)
// ---------------------------------------------------------------------------
#define SA_STRIDE 40
#define SB_STRIDE 34

__device__ inline void stage_a64x32(const _Float16* __restrict__ A,
                                    int row_stride_halves, int tid,
                                    _Float16 (*sA)[SA_STRIDE]) {
    // 64 rows x 32 halves = 256 uint4; one b128 per thread.
    const uint4* ga = (const uint4*)A;
    const int r = tid >> 2, q = tid & 3;
    const uint4 d = ga[r * (row_stride_halves / 8) + q];
    *(uint4*)&sA[r][q * 8] = d;
}

__device__ inline void stage_b32x256_t(const _Float16* __restrict__ Bw,
                                       int row_stride_halves, int tid,
                                       _Float16 (*sB)[SB_STRIDE]) {
    // 32 rows(k) x 256 cols(n) = 1024 uint4; 4 b128 per thread, transposed.
    const uint4* gb = (const uint4*)Bw;
    unsigned short* sb = (unsigned short*)&sB[0][0];
#pragma unroll
    for (int i = 0; i < 4; ++i) {
        const int idx = tid + i * 256;
        const int r = idx >> 5, cu = idx & 31;   // cu: uint4 col (8 halves)
        const uint4 d = gb[r * (row_stride_halves / 8) + cu];
        const unsigned int w[4] = {d.x, d.y, d.z, d.w};
        const int nbase = cu * 8;
#pragma unroll
        for (int t = 0; t < 4; ++t) {
            sb[(nbase + 2 * t)     * SB_STRIDE + r] = (unsigned short)(w[t] & 0xffffu);
            sb[(nbase + 2 * t + 1) * SB_STRIDE + r] = (unsigned short)(w[t] >> 16);
        }
    }
}

// ---------------------------------------------------------------------------
// Kernel 3: QKV GEMM (8192x1024 x 1024x3072), WMMA f16, scatter to q/k/v
//           layout [b][h][t][hd].
//           Block tile 64M x 256N, 8 waves as 2Mx4N; wave tile 32M x 64N
//           (2 A-frags x 4 B-frags -> 8 WMMAs per 32-deep K slab).
// ---------------------------------------------------------------------------
__global__ __launch_bounds__(256) void qkv_gemm(
    const _Float16* __restrict__ A, const _Float16* __restrict__ Bw,
    const float* __restrict__ bias,
    _Float16* __restrict__ qo, _Float16* __restrict__ ko,
    _Float16* __restrict__ vo) {
    constexpr int Kd = D_, Nd = N3D_;
    __shared__ _Float16 sA[64][SA_STRIDE];
    __shared__ _Float16 sB[256][SB_STRIDE];

    const int tid = threadIdx.x, wid = tid >> 5, lane = tid & 31;
    const int m0 = blockIdx.x * 64, n0 = blockIdx.y * 256;
    const int mo = (wid >> 2) * 32, no = (wid & 3) * 64;

    const v8f vzero = {};
    v8f c[2][4];
#pragma unroll
    for (int i = 0; i < 2; ++i)
#pragma unroll
        for (int j = 0; j < 4; ++j) c[i][j] = vzero;

    for (int k0 = 0; k0 < Kd; k0 += 32) {
        __syncthreads();
        stage_a64x32(A + (size_t)m0 * Kd + k0, Kd, tid, sA);
        stage_b32x256_t(Bw + (size_t)k0 * Nd + n0, Nd, tid, sB);
        if (k0 + 32 < Kd) {
            __builtin_prefetch(A + (size_t)(m0 + (tid & 63)) * Kd + k0 + 32, 0, 0);
            __builtin_prefetch(Bw + (size_t)(k0 + 32 + (tid & 31)) * Nd + n0, 0, 0);
        }
        __syncthreads();

        const v16h a0 = load_frag_a16(&sA[mo][0], SA_STRIDE, lane);
        const v16h a1 = load_frag_a16(&sA[mo + 16][0], SA_STRIDE, lane);
#pragma unroll
        for (int j = 0; j < 4; ++j) {
            const v16h b = load_frag_b16(&sB[no + j * 16][0], SB_STRIDE, 1, lane);
            c[0][j] = wmma_f16(a0, b, c[0][j]);
            c[1][j] = wmma_f16(a1, b, c[1][j]);
        }
    }

    // Epilogue: scatter into q/k/v [b][h][t][hd]
    const int mb = (lane & 16) ? 8 : 0;
#pragma unroll
    for (int i = 0; i < 2; ++i)
#pragma unroll
        for (int r = 0; r < 8; ++r) {
            const int gm = m0 + mo + i * 16 + mb + r;
            const int bb = gm >> 9, tt = gm & 511;
#pragma unroll
            for (int j = 0; j < 4; ++j) {
                const int gn = n0 + no + j * 16 + (lane & 15);
                const float val = c[i][j][r] + bias[gn];
                const int s   = gn >> 10;
                const int rem = gn & 1023;
                const int hh  = rem >> 6, dd = rem & 63;
                _Float16* dst = (s == 0) ? qo : (s == 1) ? ko : vo;
                dst[(((size_t)bb * H_ + hh) * T_ + tt) * HD_ + dd] = (_Float16)val;
            }
        }
}

// ---------------------------------------------------------------------------
// Kernel 4: attention.  grid = (T/16, B*H).  Block = 256 threads (8 waves).
//   Phase 1: S[16][512] = QK^T/8 + rel_bias, causal mask        (WMMA)
//   Phase 2: row softmax + entropy (exact clip(p,1e-9) form)
//   Phase 3: O[16][64] = P * V, split-K across waves            (WMMA)
// ---------------------------------------------------------------------------
__global__ __launch_bounds__(256) void attn_kernel(
    const _Float16* __restrict__ qm, const _Float16* __restrict__ km,
    const _Float16* __restrict__ vm, const float* __restrict__ rel_bias,
    float* __restrict__ ent_part, _Float16* __restrict__ ao) {
    __shared__ float S[16][520];

    const int tid = threadIdx.x, wid = tid >> 5, lane = tid & 31;
    const int bh = blockIdx.y, bb = bh >> 4, hh = bh & 15;
    const int qt0 = blockIdx.x << 4;

    const _Float16* qb = qm + ((size_t)bh * T_ + qt0) * HD_;
    const _Float16* kb = km + (size_t)bh * T_ * HD_;
    const _Float16* vb = vm + (size_t)bh * T_ * HD_;
    const float*    rb = rel_bias + hh * RELW;

    // Q fragments (hd 0..31 and 32..63) shared by all score tiles.
    const v16h aq0 = load_frag_a16(qb, HD_, lane);
    const v16h aq1 = load_frag_a16(qb + 32, HD_, lane);

    // ---- Phase 1: scores ----
    for (int nt = wid; nt < 32; nt += 8) {
        const int tk0 = nt << 4;
        const v16h bk0 = load_frag_b16(kb + (size_t)tk0 * HD_,      HD_, 1, lane);
        const v16h bk1 = load_frag_b16(kb + (size_t)tk0 * HD_ + 32, HD_, 1, lane);
        v8f acc = {};
        acc = wmma_f16(aq0, bk0, acc);
        acc = wmma_f16(aq1, bk1, acc);

        const int mbase = (lane & 16) ? 8 : 0;
        const int tk = tk0 + (lane & 15);
#pragma unroll
        for (int r = 0; r < 8; ++r) {
            const int mrow = mbase + r;
            const int tq = qt0 + mrow;
            float sv = acc[r] * 0.125f + rb[tk - tq + (T_ - 1)];
            if (tk > tq) sv = -1e30f;
            S[mrow][tk] = sv;
        }
    }
    __syncthreads();

    // ---- Phase 2: softmax + entropy (2 rows per wave) ----
#pragma unroll
    for (int rr = 0; rr < 2; ++rr) {
        const int row = wid * 2 + rr;
        const int tq = qt0 + row;
        float m = -3.0e38f;
        for (int c = lane; c < T_; c += 32) m = fmaxf(m, S[row][c]);
        m = wave_max(m);
        float z = 0.f;
        for (int c = lane; c < T_; c += 32) z += __expf(S[row][c] - m);
        z = wave_sum(z);
        const float inv = 1.f / z;
        float ent = 0.f;
        for (int c = lane; c < T_; c += 32) {
            const float p = __expf(S[row][c] - m) * inv;
            S[row][c] = p;
            const float pc = fmaxf(p, 1e-9f);
            ent -= pc * __logf(pc);
        }
        ent = wave_sum(ent);
        if (lane == 0) ent_part[(size_t)bh * T_ + tq] = ent;
    }
    __syncthreads();

    // ---- Phase 3: O = P * V, split-K (waves 0-3: tk<256, waves 4-7: tk>=256)
    const int ntv   = wid & 3;   // hd tile (16 wide)
    const int khalf = wid >> 2;  // K half
    v8f oc = {};
    for (int it = 0; it < 8; ++it) {
        const int tk0 = khalf * 256 + it * 32;
        const v16h ap = load_frag_a_f32(&S[0][tk0], 520, lane);
        const v16h bv = load_frag_b16(vb + (size_t)tk0 * HD_ + ntv * 16,
                                      1, HD_, lane);
        oc = wmma_f16(ap, bv, oc);
    }
    __syncthreads();
    float* part = &S[0][0];   // reuse LDS (P fully consumed): 4*8*32 floats
    if (khalf == 1) {
#pragma unroll
        for (int r = 0; r < 8; ++r) part[(ntv * 8 + r) * 32 + lane] = oc[r];
    }
    __syncthreads();
    if (khalf == 0) {
        const int mbase = (lane & 16) ? 8 : 0;
#pragma unroll
        for (int r = 0; r < 8; ++r) {
            const float val = oc[r] + part[(ntv * 8 + r) * 32 + lane];
            const int tt = qt0 + mbase + r;
            const int hd = ntv * 16 + (lane & 15);
            ao[((size_t)bb * T_ + tt) * D_ + hh * HD_ + hd] = (_Float16)val;
        }
    }
}

// ---------------------------------------------------------------------------
// Kernel 5: output projection (8192x1024 x 1024x1024) + bias + residual.
//           Same 64M x 256N block tile as qkv_gemm.
// ---------------------------------------------------------------------------
__global__ __launch_bounds__(256) void out_gemm(
    const _Float16* __restrict__ A, const _Float16* __restrict__ Bw,
    const float* __restrict__ bias, const float* __restrict__ resid,
    float* __restrict__ out) {
    constexpr int Kd = D_, Nd = D_;
    __shared__ _Float16 sA[64][SA_STRIDE];
    __shared__ _Float16 sB[256][SB_STRIDE];

    const int tid = threadIdx.x, wid = tid >> 5, lane = tid & 31;
    const int m0 = blockIdx.x * 64, n0 = blockIdx.y * 256;
    const int mo = (wid >> 2) * 32, no = (wid & 3) * 64;

    const v8f vzero = {};
    v8f c[2][4];
#pragma unroll
    for (int i = 0; i < 2; ++i)
#pragma unroll
        for (int j = 0; j < 4; ++j) c[i][j] = vzero;

    for (int k0 = 0; k0 < Kd; k0 += 32) {
        __syncthreads();
        stage_a64x32(A + (size_t)m0 * Kd + k0, Kd, tid, sA);
        stage_b32x256_t(Bw + (size_t)k0 * Nd + n0, Nd, tid, sB);
        if (k0 + 32 < Kd) {
            __builtin_prefetch(A + (size_t)(m0 + (tid & 63)) * Kd + k0 + 32, 0, 0);
            __builtin_prefetch(Bw + (size_t)(k0 + 32 + (tid & 31)) * Nd + n0, 0, 0);
        }
        __syncthreads();

        const v16h a0 = load_frag_a16(&sA[mo][0], SA_STRIDE, lane);
        const v16h a1 = load_frag_a16(&sA[mo + 16][0], SA_STRIDE, lane);
#pragma unroll
        for (int j = 0; j < 4; ++j) {
            const v16h b = load_frag_b16(&sB[no + j * 16][0], SB_STRIDE, 1, lane);
            c[0][j] = wmma_f16(a0, b, c[0][j]);
            c[1][j] = wmma_f16(a1, b, c[1][j]);
        }
    }

    const int mb = (lane & 16) ? 8 : 0;
#pragma unroll
    for (int i = 0; i < 2; ++i)
#pragma unroll
        for (int r = 0; r < 8; ++r) {
            const size_t gm = (size_t)(m0 + mo + i * 16 + mb + r);
#pragma unroll
            for (int j = 0; j < 4; ++j) {
                const int gn = n0 + no + j * 16 + (lane & 15);
                const float val = c[i][j][r] + bias[gn];
                out[gm * Nd + gn] = val + resid[gm * Nd + gn];
            }
        }
}

// ---------------------------------------------------------------------------
// Kernel 6: entropy finalize — mean over heads, append after main output.
// ---------------------------------------------------------------------------
__global__ __launch_bounds__(256) void ent_finalize(
    const float* __restrict__ ent_part, float* __restrict__ out) {
    const int i = blockIdx.x * 256 + threadIdx.x;   // 0..8191  (b*512 + t)
    const int bb = i >> 9, tt = i & 511;
    float s = 0.f;
#pragma unroll
    for (int hh = 0; hh < H_; ++hh)
        s += ent_part[((size_t)(bb * H_ + hh)) * T_ + tt];
    out[OUT_ELEMS + i] = s * (1.f / H_);
}

// ---------------------------------------------------------------------------
// Launch
// ---------------------------------------------------------------------------
extern "C" void kernel_launch(void* const* d_in, const int* in_sizes, int n_in,
                              void* d_out, int out_size, void* d_ws,
                              size_t ws_size, hipStream_t stream) {
    const float* x        = (const float*)d_in[0];
    const float* qkv_w    = (const float*)d_in[1];
    const float* qkv_b    = (const float*)d_in[2];
    const float* out_w    = (const float*)d_in[3];
    const float* out_b    = (const float*)d_in[4];
    const float* ln_g     = (const float*)d_in[5];
    const float* ln_b     = (const float*)d_in[6];
    const float* rel_bias = (const float*)d_in[7];
    float* out = (float*)d_out;

    char* ws = (char*)d_ws;
    size_t off = 0;
    _Float16* xn      = (_Float16*)(ws + off); off += (size_t)B_ * T_ * D_ * 2;   // 16MB
    _Float16* qkvw_h  = (_Float16*)(ws + off); off += (size_t)D_ * N3D_ * 2;      // 6MB
    _Float16* outw_h  = (_Float16*)(ws + off); off += (size_t)D_ * D_ * 2;        // 2MB
    _Float16* qh      = (_Float16*)(ws + off); off += (size_t)B_ * T_ * D_ * 2;   // 16MB
    _Float16* kh      = (_Float16*)(ws + off); off += (size_t)B_ * T_ * D_ * 2;   // 16MB
    _Float16* vh      = (_Float16*)(ws + off); off += (size_t)B_ * T_ * D_ * 2;   // 16MB
    _Float16* ao      = (_Float16*)(ws + off); off += (size_t)B_ * T_ * D_ * 2;   // 16MB
    float*    entp    = (float*)(ws + off);    off += (size_t)B_ * H_ * T_ * 4;   // 512KB

    prep_kernel<<<(D_ * N3D_ / 4) / 256, 256, 0, stream>>>(
        (const float4*)qkv_w, (const float4*)out_w, qkvw_h, outw_h);
    ln_kernel<<<B_ * T_, 256, 0, stream>>>(x, ln_g, ln_b, xn);
    qkv_gemm<<<dim3((B_ * T_) / 64, N3D_ / 256), 256, 0, stream>>>(
        xn, qkvw_h, qkv_b, qh, kh, vh);
    attn_kernel<<<dim3(T_ / 16, B_ * H_), 256, 0, stream>>>(
        qh, kh, vh, rel_bias, entp, ao);
    out_gemm<<<dim3((B_ * T_) / 64, D_ / 256), 256, 0, stream>>>(
        ao, outw_h, out_b, x, out);
    ent_finalize<<<(B_ * T_) / 256, 256, 0, stream>>>(entp, out);
}
